// TransitionModel_9594956939411
// MI455X (gfx1250) — matrix operand order repaired and checked by reading, not compiled, not run on hardware
//
#include <hip/hip_runtime.h>

typedef __attribute__((ext_vector_type(16))) __bf16 v16bf;
typedef __attribute__((ext_vector_type(8)))  __bf16 v8bf;
typedef __attribute__((ext_vector_type(8)))  float  v8f;
typedef __bf16 bf16;

#define B_   4
#define T_   8
#define P_   196
#define N_   (T_*P_)          // 1568
#define EMB_ 384
#define PE_  32
#define AE_  32
#define PD_  7
#define AD_  7
#define D_   448
#define H_   8
#define DH_  64
#define I_   512
#define M_   1792
#define L_   4
#define ROWS_ (B_*N_)         // 6272
#define QKV_  (3*I_)          // 1536
#define SCALE_ 0.125f
#define EPS_  1e-5f

// ------------------------------------------------------------- CDNA5 async
// global -> LDS async copy (16B per lane), tracked by ASYNCcnt.
__device__ __forceinline__ void async_ld_b128(const void* g, void* lds) {
  unsigned l = (unsigned)(unsigned long long)lds;   // addr[31:0] == LDS offset
  asm volatile("global_load_async_to_lds_b128 %0, %1, off"
               :: "v"(l), "v"((unsigned long long)g) : "memory");
}
__device__ __forceinline__ void wait_async() {
  asm volatile("s_wait_asynccnt 0" ::: "memory");
}

__device__ __forceinline__ v16bf frag16(v8bf lo, v8bf hi) {
  v16bf r;
#pragma unroll
  for (int i = 0; i < 8; i++) { r[i] = lo[i]; r[8 + i] = hi[i]; }
  return r;
}

// ---------------------------------------------------------------- utilities
__global__ __launch_bounds__(256) void cvt_bf16_k(const float* __restrict__ s,
                                                  bf16* __restrict__ d, int n) {
  int i = blockIdx.x * 256 + threadIdx.x;
  if (i < n) d[i] = (bf16)s[i];
}

__device__ __forceinline__ float gelu_exact(float x) {
  return 0.5f * x * (1.0f + erff(x * 0.70710678118654752f));
}

// --------------------------------------------------------------- embedding
__global__ __launch_bounds__(256) void build_x_k(
    const float* __restrict__ latent, const float* __restrict__ proprio,
    const float* __restrict__ action, const float* __restrict__ wp,
    const float* __restrict__ bp, const float* __restrict__ wa,
    const float* __restrict__ ba, const float* __restrict__ pos,
    float* __restrict__ x) {
  int idx = blockIdx.x * 256 + threadIdx.x;
  if (idx >= ROWS_ * D_) return;
  int d = idx % D_;
  int row = idx / D_;
  int b = row / N_, n = row % N_;
  int t = n / P_, p = n % P_;
  float v;
  if (d < EMB_) {
    v = latent[(((size_t)(b * T_ + t)) * P_ + p) * EMB_ + d];
  } else if (d < EMB_ + PE_) {
    int f = d - EMB_;
    float acc = bp[f];
#pragma unroll
    for (int k = 0; k < PD_; k++) acc += proprio[(b * T_ + t) * PD_ + k] * wp[k * PE_ + f];
    v = acc;
  } else {
    int f = d - EMB_ - PE_;
    float acc = ba[f];
#pragma unroll
    for (int k = 0; k < AD_; k++) acc += action[(b * T_ + t) * AD_ + k] * wa[k * AE_ + f];
    v = acc;
  }
  x[idx] = v + pos[(size_t)n * D_ + d];
}

// --------------------------------------------------------------- layernorm
__global__ __launch_bounds__(128) void layernorm_k(
    const float* __restrict__ x, const float* __restrict__ g,
    const float* __restrict__ be, bf16* __restrict__ out_bf,
    float* __restrict__ out_f) {
  int row = blockIdx.x;
  const float* xr = x + (size_t)row * D_;
  float s = 0.f, s2 = 0.f;
  for (int d = threadIdx.x; d < D_; d += 128) {
    float v = xr[d];
    s += v; s2 += v * v;
  }
#pragma unroll
  for (int off = 16; off; off >>= 1) {
    s  += __shfl_xor(s, off);
    s2 += __shfl_xor(s2, off);
  }
  __shared__ float rs[4], rs2[4];
  int wid = threadIdx.x >> 5, lane = threadIdx.x & 31;
  if (lane == 0) { rs[wid] = s; rs2[wid] = s2; }
  __syncthreads();
  s  = rs[0] + rs[1] + rs[2] + rs[3];
  s2 = rs2[0] + rs2[1] + rs2[2] + rs2[3];
  float mean = s * (1.0f / D_);
  float var  = s2 * (1.0f / D_) - mean * mean;
  float inv  = rsqrtf(var + EPS_);
  for (int d = threadIdx.x; d < D_; d += 128) {
    float v = (xr[d] - mean) * inv * g[d] + be[d];
    if (out_f)  out_f[(size_t)row * D_ + d]  = v;
    if (out_bf) out_bf[(size_t)row * D_ + d] = (bf16)v;
  }
}

// ------------------------------------------------------------- V transpose
// vT[b,h,dh,n] = qkv[b,n, 2I + h*64+dh]; coalesced reads, strided writes.
__global__ __launch_bounds__(256) void vtrans_k(const bf16* __restrict__ qkv,
                                                bf16* __restrict__ vT) {
  int idx = blockIdx.x * 256 + threadIdx.x;
  if (idx >= ROWS_ * I_) return;
  int c = idx % I_;
  int row = idx / I_;
  int b = row / N_, n = row % N_;
  int h = c >> 6, dh = c & 63;
  vT[(((size_t)(b * H_ + h)) * DH_ + dh) * N_ + n] =
      qkv[(size_t)row * QKV_ + 2 * I_ + c];
}

// ------------------------------------------------------------------- GEMM
// C[M x Nc] = A[M x K](bf16 rm) * W[K x Nc](bf16 rm); epilogue:
//   (+bias) -> (GELU) -> (+resid f32) -> f32 and/or bf16 stores.
// Block tile 128x64, 8 waves (4x2), each wave 32x32 via 2x2 wmma tiles.
// Double-buffered LDS: A tiles via global_load_async_to_lds_b128 (ASYNCcnt),
// W tiles prefetched to registers and stored transposed; one barrier/K-step.
#define BM 128
#define BN 64
#define BK 32
#define ALD 40   // padded A stride (bf16): 80B rows, 16B-aligned
#define WLD 40   // padded W^T stride (bf16): 80B rows, 16B-aligned

__global__ __launch_bounds__(256) void gemm_bf16_k(
    const bf16* __restrict__ A, const bf16* __restrict__ W,
    const float* __restrict__ bias, const float* __restrict__ resid,
    float* __restrict__ outF, bf16* __restrict__ outB,
    int K, int Nc, int act) {
  __shared__ bf16 As[2][BM][ALD];    // 2 x row-major A tile
  __shared__ bf16 WsT[2][BN][WLD];   // 2 x transposed W tile: [col][k]

  int tid = threadIdx.x;
  int wid = tid >> 5, lane = tid & 31;
  int wm = wid >> 1, wn = wid & 1;
  int half = lane >> 4, ln16 = lane & 15;
  int rowBase = blockIdx.y * BM;
  int colBase = blockIdx.x * BN;

  // per-thread staging coordinates
  int sr = tid >> 1,  sc = (tid & 1) * 16;   // A: 16 bf16 / thread
  int wr = tid >> 3,  wc = (tid & 7) * 8;    // W: 8 bf16 / thread
  const bf16* aSrc = A + (size_t)(rowBase + sr) * K + sc;
  const bf16* wSrc = W + (size_t)wr * Nc + colBase + wc;

  v8f acc[2][2] = {};
  int nt = K / BK;

  // prologue: stage tile 0
  async_ld_b128(aSrc,     &As[0][sr][sc]);
  async_ld_b128(aSrc + 8, &As[0][sr][sc + 8]);
  {
    v8bf w8 = *(const v8bf*)wSrc;
#pragma unroll
    for (int j = 0; j < 8; j++) WsT[0][wc + j][wr] = w8[j];
  }
  wait_async();
  __syncthreads();

  for (int i = 0; i < nt; i++) {
    int cur = i & 1, nxt = cur ^ 1;
    v8bf w8 = {};
    if (i + 1 < nt) {  // kick off tile i+1 while computing tile i
      const bf16* aS = aSrc + (i + 1) * BK;
      async_ld_b128(aS,     &As[nxt][sr][sc]);
      async_ld_b128(aS + 8, &As[nxt][sr][sc + 8]);
      w8 = *(const v8bf*)(wSrc + (size_t)(i + 1) * BK * Nc);
    }

    // B fragments: lane = col, k = half*16 .. +15 contiguous -> 2x b128
    v16bf bfr[2];
#pragma unroll
    for (int ni = 0; ni < 2; ni++) {
      int bc = wn * 32 + ni * 16 + ln16;
      bfr[ni] = frag16(*(const v8bf*)&WsT[cur][bc][half * 16],
                       *(const v8bf*)&WsT[cur][bc][half * 16 + 8]);
    }
#pragma unroll
    for (int mi = 0; mi < 2; mi++) {
      // A fragment: lane = row, k = half*16 .. +15 contiguous -> 2x b128
      int ar = wm * 32 + mi * 16 + ln16;
      v16bf a = frag16(*(const v8bf*)&As[cur][ar][half * 16],
                       *(const v8bf*)&As[cur][ar][half * 16 + 8]);
#pragma unroll
      for (int ni = 0; ni < 2; ni++)
        acc[mi][ni] = __builtin_amdgcn_wmma_f32_16x16x32_bf16(
            false, a, false, bfr[ni], (short)0, acc[mi][ni], false, false);
    }

    if (i + 1 < nt) {
#pragma unroll
      for (int j = 0; j < 8; j++) WsT[nxt][wc + j][wr] = w8[j];
      wait_async();
    }
    __syncthreads();
  }

  // epilogue: C layout row = 8*half+v, col = ln16; uniform branches hoisted
  int gr0[2], gc0[2];
#pragma unroll
  for (int mi = 0; mi < 2; mi++) gr0[mi] = rowBase + wm * 32 + mi * 16 + half * 8;
#pragma unroll
  for (int ni = 0; ni < 2; ni++) gc0[ni] = colBase + wn * 32 + ni * 16 + ln16;

  if (bias) {
#pragma unroll
    for (int ni = 0; ni < 2; ni++) {
      float bv = bias[gc0[ni]];
#pragma unroll
      for (int mi = 0; mi < 2; mi++)
#pragma unroll
        for (int v = 0; v < 8; v++) acc[mi][ni][v] += bv;
    }
  }
  if (act) {
#pragma unroll
    for (int mi = 0; mi < 2; mi++)
#pragma unroll
      for (int ni = 0; ni < 2; ni++)
#pragma unroll
        for (int v = 0; v < 8; v++) acc[mi][ni][v] = gelu_exact(acc[mi][ni][v]);
  }
  if (resid) {
#pragma unroll
    for (int mi = 0; mi < 2; mi++)
#pragma unroll
      for (int ni = 0; ni < 2; ni++)
#pragma unroll
        for (int v = 0; v < 8; v++)
          acc[mi][ni][v] += resid[(size_t)(gr0[mi] + v) * Nc + gc0[ni]];
  }
  if (outF) {
#pragma unroll
    for (int mi = 0; mi < 2; mi++)
#pragma unroll
      for (int ni = 0; ni < 2; ni++)
#pragma unroll
        for (int v = 0; v < 8; v++)
          outF[(size_t)(gr0[mi] + v) * Nc + gc0[ni]] = acc[mi][ni][v];
  }
  if (outB) {
#pragma unroll
    for (int mi = 0; mi < 2; mi++)
#pragma unroll
      for (int ni = 0; ni < 2; ni++)
#pragma unroll
        for (int v = 0; v < 8; v++)
          outB[(size_t)(gr0[mi] + v) * Nc + gc0[ni]] = (bf16)acc[mi][ni][v];
  }
}

// -------------------------------------------------------- flash attention
// grid (N_/32, H_, B_), block 64 (2 waves); wave handles 16 query rows.
__global__ __launch_bounds__(64) void attention_k(const bf16* __restrict__ qkv,
                                                  const bf16* __restrict__ vT,
                                                  bf16* __restrict__ o) {
  __shared__ bf16 Pld[2][16][32];
  int wid = threadIdx.x >> 5, lane = threadIdx.x & 31;
  int half = lane >> 4, ln16 = lane & 15;
  int b = blockIdx.z, h = blockIdx.y;
  int qbase = blockIdx.x * 32 + wid * 16;

  const bf16* qk = qkv + (size_t)b * N_ * QKV_ + h * DH_;
  const bf16* vh = vT + ((size_t)(b * H_ + h)) * DH_ * N_;

  // Q fragments: 16x64 = two 16x32 A-frags (contiguous b128 pairs)
  v16bf qf[2];
  {
    const bf16* qr = qk + (size_t)(qbase + ln16) * QKV_;
#pragma unroll
    for (int kk = 0; kk < 2; kk++)
      qf[kk] = frag16(*(const v8bf*)(qr + kk * 32 + half * 16),
                      *(const v8bf*)(qr + kk * 32 + half * 16 + 8));
  }

  v8f oacc[4] = {};
  float mrow[8], lrow[8];
  int rowFrame[8];
#pragma unroll
  for (int v = 0; v < 8; v++) {
    mrow[v] = -INFINITY; lrow[v] = 0.f;
    rowFrame[v] = (qbase + half * 8 + v) / P_;
  }
  int kend = ((blockIdx.x * 32 + 31) / P_ + 1) * P_;  // block-uniform

  for (int kt = 0; kt < kend; kt += 32) {
    // S = Q K^T for a 32-key tile: K frag contiguous (lane = key row)
    v8f s[2] = {};
#pragma unroll
    for (int ch = 0; ch < 2; ch++) {
      const bf16* kr = qk + (size_t)(kt + ch * 16 + ln16) * QKV_ + I_;
#pragma unroll
      for (int kk = 0; kk < 2; kk++) {
        v16bf kf = frag16(*(const v8bf*)(kr + kk * 32 + half * 16),
                          *(const v8bf*)(kr + kk * 32 + half * 16 + 8));
        s[ch] = __builtin_amdgcn_wmma_f32_16x16x32_bf16(
            false, qf[kk], false, kf, (short)0, s[ch], false, false);
      }
    }
    // scale + frame mask + online softmax
    int f0 = (kt + ln16) / P_;
    int f1 = (kt + 16 + ln16) / P_;
#pragma unroll
    for (int v = 0; v < 8; v++) {
      float a0 = (f0 <= rowFrame[v]) ? s[0][v] * SCALE_ : -INFINITY;
      float a1 = (f1 <= rowFrame[v]) ? s[1][v] * SCALE_ : -INFINITY;
      float mx = fmaxf(a0, a1);
#pragma unroll
      for (int off = 8; off; off >>= 1) mx = fmaxf(mx, __shfl_xor(mx, off));
      float mnew = fmaxf(mrow[v], mx);
      float al = __expf(mrow[v] - mnew);
      mrow[v] = mnew;
      float p0 = __expf(a0 - mnew);
      float p1 = __expf(a1 - mnew);
      float rsum = p0 + p1;
#pragma unroll
      for (int off = 8; off; off >>= 1) rsum += __shfl_xor(rsum, off);
      lrow[v] = lrow[v] * al + rsum;
#pragma unroll
      for (int cb = 0; cb < 4; cb++) oacc[cb][v] *= al;
      Pld[wid][half * 8 + v][ln16]      = (bf16)p0;
      Pld[wid][half * 8 + v][16 + ln16] = (bf16)p1;
    }
    __syncthreads();
    // P (A layout, contiguous b128 pair) @ V (B frags from vT, contiguous)
    v16bf pf = frag16(*(const v8bf*)&Pld[wid][ln16][half * 16],
                      *(const v8bf*)&Pld[wid][ln16][half * 16 + 8]);
#pragma unroll
    for (int cb = 0; cb < 4; cb++) {
      const bf16* vr = vh + (size_t)(cb * 16 + ln16) * N_ + kt + half * 16;
      v16bf vf = frag16(*(const v8bf*)vr, *(const v8bf*)(vr + 8));
      oacc[cb] = __builtin_amdgcn_wmma_f32_16x16x32_bf16(
          false, pf, false, vf, (short)0, oacc[cb], false, false);
    }
    __syncthreads();
  }

  // O /= l, write bf16 in (B,N,H*DH) layout
#pragma unroll
  for (int v = 0; v < 8; v++) {
    float linv = 1.0f / lrow[v];
    int row = qbase + half * 8 + v;
    size_t base = (size_t)(b * N_ + row) * I_ + h * DH_;
#pragma unroll
    for (int cb = 0; cb < 4; cb++)
      o[base + cb * 16 + ln16] = (bf16)(oacc[cb][v] * linv);
  }
}

// ------------------------------------------------------------------- host
extern "C" void kernel_launch(void* const* d_in, const int* in_sizes, int n_in,
                              void* d_out, int out_size, void* d_ws, size_t ws_size,
                              hipStream_t stream) {
  const float* latent  = (const float*)d_in[0];
  const float* proprio = (const float*)d_in[1];
  const float* action  = (const float*)d_in[2];
  const float* wp      = (const float*)d_in[3];
  const float* bp      = (const float*)d_in[4];
  const float* wa      = (const float*)d_in[5];
  const float* ba      = (const float*)d_in[6];
  const float* pos     = (const float*)d_in[7];
  const float* ln1_g   = (const float*)d_in[8];
  const float* ln1_b   = (const float*)d_in[9];
  const float* wqkv    = (const float*)d_in[10];
  const float* wo      = (const float*)d_in[11];
  const float* bo      = (const float*)d_in[12];
  const float* ln2_g   = (const float*)d_in[13];
  const float* ln2_b   = (const float*)d_in[14];
  const float* w1      = (const float*)d_in[15];
  const float* b1      = (const float*)d_in[16];
  const float* w2      = (const float*)d_in[17];
  const float* b2      = (const float*)d_in[18];
  const float* fin_g   = (const float*)d_in[19];
  const float* fin_b   = (const float*)d_in[20];

  char* ws = (char*)d_ws;
  size_t off = 0;
  auto alloc = [&](size_t bytes) -> void* {
    void* p = ws + off;
    off += (bytes + 255) & ~(size_t)255;
    return p;
  };
  float* x     = (float*)alloc((size_t)ROWS_ * D_ * 4);
  bf16* hbuf   = (bf16*)alloc((size_t)ROWS_ * D_ * 2);
  bf16* qkvb   = (bf16*)alloc((size_t)ROWS_ * QKV_ * 2);
  bf16* obuf   = (bf16*)alloc((size_t)ROWS_ * I_ * 2);
  bf16* vtb    = (bf16*)alloc((size_t)ROWS_ * I_ * 2);
  bf16* gbuf   = (bf16*)alloc((size_t)ROWS_ * M_ * 2);
  bf16* wqkv_b = (bf16*)alloc((size_t)L_ * D_ * QKV_ * 2);
  bf16* wo_b   = (bf16*)alloc((size_t)L_ * I_ * D_ * 2);
  bf16* w1_b   = (bf16*)alloc((size_t)L_ * D_ * M_ * 2);
  bf16* w2_b   = (bf16*)alloc((size_t)L_ * M_ * D_ * 2);

  auto cvt = [&](const float* s, bf16* d, int n) {
    cvt_bf16_k<<<(n + 255) / 256, 256, 0, stream>>>(s, d, n);
  };
  cvt(wqkv, wqkv_b, L_ * D_ * QKV_);
  cvt(wo,   wo_b,   L_ * I_ * D_);
  cvt(w1,   w1_b,   L_ * D_ * M_);
  cvt(w2,   w2_b,   L_ * M_ * D_);

  build_x_k<<<(ROWS_ * D_ + 255) / 256, 256, 0, stream>>>(
      latent, proprio, action, wp, bp, wa, ba, pos, x);

  for (int l = 0; l < L_; l++) {
    layernorm_k<<<ROWS_, 128, 0, stream>>>(x, ln1_g + l * D_, ln1_b + l * D_,
                                           hbuf, nullptr);
    gemm_bf16_k<<<dim3(QKV_ / BN, ROWS_ / BM), 256, 0, stream>>>(
        hbuf, wqkv_b + (size_t)l * D_ * QKV_, nullptr, nullptr, nullptr, qkvb,
        D_, QKV_, 0);
    vtrans_k<<<(ROWS_ * I_ + 255) / 256, 256, 0, stream>>>(qkvb, vtb);
    attention_k<<<dim3(N_ / 32, H_, B_), 64, 0, stream>>>(qkvb, vtb, obuf);
    gemm_bf16_k<<<dim3(D_ / BN, ROWS_ / BM), 256, 0, stream>>>(
        obuf, wo_b + (size_t)l * I_ * D_, bo + l * D_, x, x, nullptr,
        I_, D_, 0);
    layernorm_k<<<ROWS_, 128, 0, stream>>>(x, ln2_g + l * D_, ln2_b + l * D_,
                                           hbuf, nullptr);
    gemm_bf16_k<<<dim3(M_ / BN, ROWS_ / BM), 256, 0, stream>>>(
        hbuf, w1_b + (size_t)l * D_ * M_, b1 + l * M_, nullptr, nullptr, gbuf,
        D_, M_, 1);
    gemm_bf16_k<<<dim3(D_ / BN, ROWS_ / BM), 256, 0, stream>>>(
        gbuf, w2_b + (size_t)l * M_ * D_, b2 + l * D_, x, x, nullptr,
        M_, D_, 0);
  }

  layernorm_k<<<ROWS_, 128, 0, stream>>>(x, fin_g, fin_b, nullptr,
                                         (float*)d_out);
}
